// MambaBeatTracker_55611236548951
// MI455X (gfx1250) — matrix operand order repaired and checked by reading, not compiled, not run on hardware
//
#include <hip/hip_runtime.h>

#define DEV __device__ __forceinline__

// ---------------- model constants ----------------
constexpr int BB = 4, TTT = 1000, FBINS = 128;
constexpr int DM = 256, DI = 512, DST = 16;
constexpr int DEPTH = 6;
constexpr float EPSV = 1e-5f;
constexpr int NTOK = BB * TTT; // 4000

// ---------------- d_in leaf indices -------------------------------------
// d_in[0] = x, then params pytree flattened with alphabetical dict keys.
enum {
  IN_X = 0,
  P_BEAT_B, P_BEAT_W,
  P_C1D_B, P_C1D_W, P_C1M_B, P_C1M_W,
  P_C2D_B, P_C2D_W, P_C2M_B, P_C2M_W,
  P_C3D_B, P_C3D_W, P_C3M_B, P_C3M_W,
  P_FUSE_B, P_FUSE_W,
  P_FUS0_B1, P_FUS0_B2, P_FUS0_LNB, P_FUS0_LNW, P_FUS0_W1, P_FUS0_W2,
  P_FUS1_B1, P_FUS1_B2, P_FUS1_LNB, P_FUS1_LNW, P_FUS1_W1, P_FUS1_W2,
  P_L0,                       // 6 layers x 17 leaves each
  P_NORMF = P_L0 + 17 * DEPTH,
  P_TEMPO_B, P_TEMPO_LNB, P_TEMPO_LNW, P_TEMPO_W
};
// per-layer leaf order (alphabetical):
enum { L_bA = 0, L_bD, L_bCB, L_bCW, L_bDTB, L_bDTW, L_bXP,
       L_fA,     L_fD, L_fCB, L_fCW, L_fDTB, L_fDTW, L_fXP,
       L_INW, L_NORMW, L_OUTW };

// ---------------- workspace layout (floats) ------------------------------
constexpr size_t O_C1   = 0;                    // 5,376,000  (aliased by BIG)
constexpr size_t O_C2   = 5376000;              // 2,560,000  (aliased by XS)
constexpr size_t O_XCAT = O_C2 + 2560000;       // 2,048,000
constexpr size_t O_H    = O_XCAT + 2048000;     // 1,024,000
constexpr size_t O_RES  = O_H + 1024000;        // 1,024,000
constexpr size_t O_XN   = O_RES + 1024000;      // 1,024,000
constexpr size_t O_DBL  = O_XN + 1024000;       //   192,000
constexpr size_t O_DTV  = O_DBL + 192000;       // 2,048,000
constexpr size_t O_YACC = O_DTV + 2048000;      // 2,048,000
constexpr size_t O_YC   = O_YACC + 2048000;     // 2,048,000
constexpr size_t O_TF   = O_YC + 2048000;       //     1,024
constexpr size_t O_TFN  = O_TF + 1024;          //     1,024

// ---------------- helpers ------------------------------------------------
DEV unsigned short f2bf(float f) {            // fp32 -> bf16, RNE
  unsigned u = __float_as_uint(f);
  unsigned r = u + 0x7FFFu + ((u >> 16) & 1u);
  return (unsigned short)(r >> 16);
}
DEV unsigned packbf(float lo, float hi) {
  return (unsigned)f2bf(lo) | ((unsigned)f2bf(hi) << 16);
}
DEV float wave_sum(float v) {
#pragma unroll
  for (int off = 16; off > 0; off >>= 1) v += __shfl_xor(v, off, 32);
  return v;
}
DEV float siluf(float x) { return x / (1.f + __expf(-x)); }

enum { ACT_NONE = 0, ACT_GELU = 1, ACT_SOFTPLUS = 2 };
DEV float act_apply(float x, int act) {
  if (act == ACT_GELU) {
    float t = tanhf(0.7978845608028654f * (x + 0.044715f * x * x * x));
    return 0.5f * x * (1.f + t);
  }
  if (act == ACT_SOFTPLUS) return (x > 20.f) ? x : log1pf(__expf(x));
  return x;
}

// ---------------- WMMA GEMM:  C[M,N] = A[M,K(lda)] * W[N,K]^T + bias -----
// Block = 128 threads (4 waves). Tile: 32(M) x 64(N). Each wave owns one
// 16-wide N sub-tile and two 16x16 accumulators (M split), sharing one B
// fragment. K staged in 64-wide slabs through double-buffered LDS (bf16).
typedef __bf16 bf16x16 __attribute__((ext_vector_type(16)));
typedef float  f32x8   __attribute__((ext_vector_type(8)));
union Frag { bf16x16 v; unsigned u[8]; };

__global__ __launch_bounds__(128)
void gemm_wmma(const float* __restrict__ A, int lda,
               const float* __restrict__ W,
               const float* __restrict__ bias,
               float* __restrict__ C,
               int M, int N, int K, int act, int accum) {
  __shared__ unsigned short lA[2][32 * 64];
  __shared__ unsigned short lW[2][64 * 64];
  const int tid  = threadIdx.x;
  const int lane = tid & 31;
  const int wave = tid >> 5;                  // 0..3 -> n sub-tile
  const int m0 = blockIdx.y * 32;
  const int n0 = blockIdx.x * 64;

  // uniform fast-path condition: whole tile in bounds, vectorizable
  const bool full = ((K & 63) == 0) && (m0 + 32 <= M) && (n0 + 64 <= N) &&
                    ((lda & 3) == 0);
  const int nstages = (K + 63) >> 6;

  auto stage = [&](int bi, int k0) {
    if (full) {
      // A tile 32x64 : 512 float4 groups, 4 per thread
#pragma unroll
      for (int j = 0; j < 4; ++j) {
        int g = tid + j * 128;
        int r = g >> 4, c4 = (g & 15) << 2;
        const float4 v = *(const float4*)(A + (size_t)(m0 + r) * lda + k0 + c4);
        unsigned* dst = (unsigned*)&lA[bi][r * 64 + c4];
        dst[0] = packbf(v.x, v.y);
        dst[1] = packbf(v.z, v.w);
      }
      // W tile 64x64 : 1024 float4 groups, 8 per thread
#pragma unroll
      for (int j = 0; j < 8; ++j) {
        int g = tid + j * 128;
        int r = g >> 4, c4 = (g & 15) << 2;
        const float4 v = *(const float4*)(W + (size_t)(n0 + r) * K + k0 + c4);
        unsigned* dst = (unsigned*)&lW[bi][r * 64 + c4];
        dst[0] = packbf(v.x, v.y);
        dst[1] = packbf(v.z, v.w);
      }
      if (k0 + 128 <= K) {                    // pull slab k0+128 toward L2
        int pm = m0 + (tid & 31), pn = n0 + (tid & 63);
        __builtin_prefetch(A + (size_t)pm * lda + k0 + 128, 0, 1);
        __builtin_prefetch(W + (size_t)pn * K + k0 + 128, 0, 1);
      }
    } else {                                  // edge tiles: guarded scalar
      for (int i = tid; i < 32 * 64; i += 128) {
        int r = i >> 6, c = i & 63;
        int gm = m0 + r, gk = k0 + c;
        float v = (gm < M && gk < K) ? A[(size_t)gm * lda + gk] : 0.f;
        lA[bi][i] = f2bf(v);
      }
      for (int i = tid; i < 64 * 64; i += 128) {
        int r = i >> 6, c = i & 63;
        int gn = n0 + r, gk = k0 + c;
        float v = (gn < N && gk < K) ? W[(size_t)gn * K + gk] : 0.f;
        lW[bi][i] = f2bf(v);
      }
    }
  };

  f32x8 acc0 = {0.f, 0.f, 0.f, 0.f, 0.f, 0.f, 0.f, 0.f};
  f32x8 acc1 = {0.f, 0.f, 0.f, 0.f, 0.f, 0.f, 0.f, 0.f};
  const int half = lane >> 4;
  const int rA   = lane & 15;
  const int rW   = (wave << 4) | (lane & 15);

  stage(0, 0);
  __syncthreads();
  for (int ks = 0; ks < nstages; ++ks) {
    const int cur = ks & 1;
    if (ks + 1 < nstages) stage(cur ^ 1, (ks + 1) << 6);
    const unsigned short* pa = lA[cur];
    const unsigned short* pw = lW[cur];
#pragma unroll
    for (int s = 0; s < 2; ++s) {
      const int kb = s * 32;
      Frag a0, a1, b;
#pragma unroll
      for (int i = 0; i < 8; ++i) {
        // A fragment K pattern (16-bit A 16x32 layout)
        int ka = kb + ((i >> 2) << 4) + (half << 3) + ((i & 3) << 1);
        a0.u[i] = *(const unsigned*)&pa[rA * 64 + ka];
        a1.u[i] = *(const unsigned*)&pa[(rA + 16) * 64 + ka];
        // B fragment K pattern (16-bit B 32x16 layout)
        int kbn = kb + ((i >> 2) << 4) + (half << 3) + ((i & 3) << 1);
        b.u[i] = *(const unsigned*)&pw[rW * 64 + kbn];
      }
      acc0 = __builtin_amdgcn_wmma_f32_16x16x32_bf16(
          false, a0.v, false, b.v, (short)0, acc0, false, false);
      acc1 = __builtin_amdgcn_wmma_f32_16x16x32_bf16(
          false, a1.v, false, b.v, (short)0, acc1, false, false);
    }
    __syncthreads();
  }

  // D layout: lane n = lane&15, rows m = v + 8*(lane>>4)
  int n = n0 + (wave << 4) + (lane & 15);
  if (n < N) {
    float bv = bias ? bias[n] : 0.f;
#pragma unroll
    for (int v = 0; v < 8; ++v) {
      int m = m0 + v + (half << 3);
      if (m < M) {
        float val = acc0[v] + bv;
        if (accum) val += C[(size_t)m * N + n];
        C[(size_t)m * N + n] = act_apply(val, act);
      }
      int m2 = m + 16;
      if (m2 < M) {
        float val = acc1[v] + bv;
        if (accum) val += C[(size_t)m2 * N + n];
        C[(size_t)m2 * N + n] = act_apply(val, act);
      }
    }
  }
}

// ---------------- CNN front-end ------------------------------------------
__global__ __launch_bounds__(256)
void conv1_kernel(const float* __restrict__ x, const float* __restrict__ w,
                  const float* __restrict__ bias, float* __restrict__ out, int ch) {
  int id = blockIdx.x * 256 + threadIdx.x;
  if (id >= BB * 32 * TTT * 42) return;
  int wi = id % 42, t = (id / 42) % TTT, c = (id / (42 * TTT)) % 32, b = id / (42 * TTT * 32);
  float wg[9];
#pragma unroll
  for (int i = 0; i < 9; ++i) wg[i] = w[c * 9 + i];
  float bv = bias[c], best = -3.0e38f;
#pragma unroll
  for (int j = 0; j < 3; ++j) {
    int u = 3 * wi + j;
    float s = bv;
#pragma unroll
    for (int kh = 0; kh < 3; ++kh) {
      int tt = t + kh - 2;                    // causal pad(2) then [:-2]
      if (tt < 0) continue;
      const float* xr = x + (((size_t)(b * TTT + tt) * FBINS + u) * 2) + ch;
#pragma unroll
      for (int kw = 0; kw < 3; ++kw) s += xr[kw * 2] * wg[kh * 3 + kw];
    }
    best = fmaxf(best, s);
  }
  out[id] = fmaxf(best, 0.f);                 // relu(maxpool3)
}

__global__ __launch_bounds__(256)
void conv2_kernel(const float* __restrict__ in, const float* __restrict__ w,
                  const float* __restrict__ bias, float* __restrict__ out) {
  int id = blockIdx.x * 256 + threadIdx.x;
  if (id >= BB * 64 * TTT * 10) return;
  int wi = id % 10, t = (id / 10) % TTT, c = (id / (10 * TTT)) % 64, b = id / (10 * TTT * 64);
  float p0 = bias[c], p1 = p0, p2 = p0;
  for (int ci = 0; ci < 32; ++ci) {
    const float* row = in + ((size_t)(b * 32 + ci) * TTT + t) * 42 + 3 * wi;
    const float* wr  = w + (c * 32 + ci) * 12;
    float xin[14];
#pragma unroll
    for (int i = 0; i < 14; ++i) xin[i] = row[i];
#pragma unroll
    for (int kw = 0; kw < 12; ++kw) {
      float wv = wr[kw];
      p0 = fmaf(xin[kw], wv, p0);
      p1 = fmaf(xin[kw + 1], wv, p1);
      p2 = fmaf(xin[kw + 2], wv, p2);
    }
  }
  out[id] = fmaxf(fmaxf(fmaxf(p0, p1), p2), 0.f);
}

__global__ __launch_bounds__(256)
void conv3_kernel(const float* __restrict__ in, const float* __restrict__ w,
                  const float* __restrict__ bias, float* __restrict__ xcat, int br) {
  int id = blockIdx.x * 256 + threadIdx.x;
  if (id >= BB * 256 * TTT) return;
  int t = id % TTT, c = (id / TTT) % 256, b = id / (TTT * 256);
  float p0 = bias[c], p1 = p0, p2 = p0;
  for (int ci = 0; ci < 64; ++ci) {
#pragma unroll
    for (int kh = 0; kh < 2; ++kh) {
      int tt = t + kh - 1;                    // pad(1) then [:-1]
      if (tt < 0 || tt >= TTT) continue;
      const float* row = in + ((size_t)(b * 64 + ci) * TTT + tt) * 10;
      const float* wr  = w + ((c * 64 + ci) * 2 + kh) * 6;
      float xin[8];
#pragma unroll
      for (int i = 0; i < 8; ++i) xin[i] = row[i];
#pragma unroll
      for (int kw = 0; kw < 6; ++kw) {
        float wv = wr[kw];
        p0 = fmaf(xin[kw], wv, p0);
        p1 = fmaf(xin[kw + 1], wv, p1);
        p2 = fmaf(xin[kw + 2], wv, p2);
      }
    }
  }
  xcat[(size_t)(b * TTT + t) * (2 * DM) + br * DM + c] =
      fmaxf(fmaxf(fmaxf(p0, p1), p2), 0.f);
}

// ---------------- norms, PE ----------------------------------------------
__global__ __launch_bounds__(128)
void ln_kernel(const float* __restrict__ in, const float* __restrict__ w,
               const float* __restrict__ b, float* __restrict__ out, int ntok) {
  int wave = threadIdx.x >> 5, lane = threadIdx.x & 31;
  int tok = blockIdx.x * 4 + wave;
  if (tok >= ntok) return;
  const float* p = in + (size_t)tok * DM;
  float v[8], s = 0.f;
#pragma unroll
  for (int i = 0; i < 8; ++i) { v[i] = p[lane + i * 32]; s += v[i]; }
  float mu = wave_sum(s) * (1.f / DM);
  float ss = 0.f;
#pragma unroll
  for (int i = 0; i < 8; ++i) { float d = v[i] - mu; ss += d * d; }
  float sc = rsqrtf(wave_sum(ss) * (1.f / DM) + EPSV);
  float* o = out + (size_t)tok * DM;
#pragma unroll
  for (int i = 0; i < 8; ++i) {
    int d = lane + i * 32;
    o[d] = (v[i] - mu) * sc * w[d] + b[d];
  }
}

// residual = first ? hidden : residual + hidden;  out = rmsnorm(residual)*w
__global__ __launch_bounds__(128)
void rmsadd_kernel(const float* __restrict__ hid, float* __restrict__ res,
                   const float* __restrict__ w, float* __restrict__ out,
                   int ntok, int first) {
  int wave = threadIdx.x >> 5, lane = threadIdx.x & 31;
  int tok = blockIdx.x * 4 + wave;
  if (tok >= ntok) return;
  const float* hr = hid + (size_t)tok * DM;
  float* rr = res + (size_t)tok * DM;
  float v[8], ss = 0.f;
#pragma unroll
  for (int i = 0; i < 8; ++i) {
    int d = lane + i * 32;
    float r = first ? hr[d] : (rr[d] + hr[d]);
    rr[d] = r; v[i] = r; ss += r * r;
  }
  float sc = rsqrtf(wave_sum(ss) * (1.f / DM) + EPSV);
  float* o = out + (size_t)tok * DM;
#pragma unroll
  for (int i = 0; i < 8; ++i) { int d = lane + i * 32; o[d] = v[i] * sc * w[d]; }
}

__global__ __launch_bounds__(256)
void posenc_kernel(float* __restrict__ h) {
  int id = blockIdx.x * 256 + threadIdx.x;
  if (id >= NTOK * DM) return;
  int d = id % DM, t = (id / DM) % TTT;
  int j = d >> 1;
  float freq = __expf(-(float)(2 * j) * (9.210340371976184f / 256.f));
  float ang = (float)t * freq;
  h[id] += (d & 1) ? __cosf(ang) : __sinf(ang);
}

// ---------------- Mamba pieces -------------------------------------------
// direction-aware causal depthwise conv (k=4) + SiLU, logical time order
__global__ __launch_bounds__(256)
void conv_silu_kernel(const float* __restrict__ xz, const float* __restrict__ cw,
                      const float* __restrict__ cb, float* __restrict__ xs, int dir) {
  int id = blockIdx.x * 256 + threadIdx.x;
  if (id >= NTOK * DI) return;
  int d = id % DI, tok = id / DI, b = tok / TTT, tau = tok % TTT;
  float wv[4];
#pragma unroll
  for (int k = 0; k < 4; ++k) wv[k] = cw[d * 4 + k];
  float s = cb[d];
#pragma unroll
  for (int k = 0; k < 4; ++k) {
    int i = tau - 3 + k;
    if (i < 0) continue;
    int tp = dir ? (TTT - 1 - i) : i;
    s = fmaf(xz[(size_t)(b * TTT + tp) * (2 * DI) + d], wv[k], s);
  }
  xs[(size_t)tok * DI + d] = siluf(s);
}

// selective scan: one thread per (b,d); T-loop internal; B/C via LDS bcast
__global__ __launch_bounds__(512)
void scan_kernel(const float* __restrict__ xs, const float* __restrict__ dbl,
                 const float* __restrict__ dtv, const float* __restrict__ A_log,
                 const float* __restrict__ Dp, float* __restrict__ yacc,
                 int dir, int add) {
  int d = threadIdx.x, b = blockIdx.x;
  float Ar[DST], h[DST];
#pragma unroll
  for (int n = 0; n < DST; ++n) { Ar[n] = -__expf(A_log[d * DST + n]); h[n] = 0.f; }
  float Dd = Dp[d];
  __shared__ float sBC[32];
  for (int tau = 0; tau < TTT; ++tau) {
    __syncthreads();
    if (threadIdx.x < 32)
      sBC[threadIdx.x] = dbl[(size_t)(b * TTT + tau) * 48 + 16 + threadIdx.x];
    __syncthreads();
    size_t o = (size_t)(b * TTT + tau) * DI + d;
    float dt = dtv[o], xv = xs[o], y = 0.f;
#pragma unroll
    for (int n = 0; n < DST; ++n) {
      float dA = __expf(dt * Ar[n]);
      h[n] = fmaf(dA, h[n], dt * sBC[n] * xv);
      y = fmaf(h[n], sBC[16 + n], y);
    }
    y = fmaf(xv, Dd, y);
    int t = dir ? (TTT - 1 - tau) : tau;
    size_t oo = (size_t)(b * TTT + t) * DI + d;
    if (add) yacc[oo] += y; else yacc[oo] = y;
  }
}

__global__ __launch_bounds__(256)
void combine_kernel(const float* __restrict__ yacc, const float* __restrict__ xz,
                    float* __restrict__ yc) {
  int id = blockIdx.x * 256 + threadIdx.x;
  if (id >= NTOK * DI) return;
  int d = id % DI, tok = id / DI;
  yc[id] = yacc[id] * siluf(xz[(size_t)tok * (2 * DI) + DI + d]);
}

__global__ __launch_bounds__(256)
void mean_kernel(const float* __restrict__ feat, float* __restrict__ tf) {
  int id = blockIdx.x * 256 + threadIdx.x;
  if (id >= BB * DM) return;
  int b = id / DM, dm = id % DM;
  float s = 0.f;
  for (int t = 0; t < TTT; ++t) s += feat[(size_t)(b * TTT + t) * DM + dm];
  tf[id] = s * (1.f / TTT);
}

// ---------------- host orchestration -------------------------------------
static inline void gemm(const float* A, int lda, const float* W, const float* bias,
                        float* C, int M, int N, int K, int act, int accum,
                        hipStream_t s) {
  dim3 g((N + 63) / 64, (M + 31) / 32);
  gemm_wmma<<<g, 128, 0, s>>>(A, lda, W, bias, C, M, N, K, act, accum);
}

extern "C" void kernel_launch(void* const* d_in, const int* in_sizes, int n_in,
                              void* d_out, int out_size, void* d_ws, size_t ws_size,
                              hipStream_t stream) {
  (void)in_sizes; (void)n_in; (void)out_size; (void)ws_size;
  const float* x = (const float*)d_in[IN_X];
  auto P = [&](int i) { return (const float*)d_in[i]; };

  float* ws   = (float*)d_ws;
  float* C1   = ws + O_C1;
  float* C2   = ws + O_C2;
  float* XCAT = ws + O_XCAT;
  float* H    = ws + O_H;
  float* RES  = ws + O_RES;
  float* XN   = ws + O_XN;
  float* DBL  = ws + O_DBL;
  float* DTV  = ws + O_DTV;
  float* YACC = ws + O_YACC;
  float* YC   = ws + O_YC;
  float* TF   = ws + O_TF;
  float* TFN  = ws + O_TFN;
  float* BIG  = ws + O_C1;   // alias: 4,096,000 <= 5,376,000 (CNN scratch dead)
  float* XS   = ws + O_C2;   // alias: 2,048,000 <= 2,560,000

  // ---- CNN front-end (mix=ch0 -> xcat[0:256), drum=ch1 -> xcat[256:512))
  for (int br = 0; br < 2; ++br) {
    conv1_kernel<<<(BB * 32 * TTT * 42 + 255) / 256, 256, 0, stream>>>(
        x, P(br ? P_C1D_W : P_C1M_W), P(br ? P_C1D_B : P_C1M_B), C1, br);
    conv2_kernel<<<(BB * 64 * TTT * 10 + 255) / 256, 256, 0, stream>>>(
        C1, P(br ? P_C2D_W : P_C2M_W), P(br ? P_C2D_B : P_C2M_B), C2);
    conv3_kernel<<<(BB * 256 * TTT + 255) / 256, 256, 0, stream>>>(
        C2, P(br ? P_C3D_W : P_C3M_W), P(br ? P_C3D_B : P_C3M_B), XCAT, br);
  }

  // ---- fuse: h = xcat @ fuse_w^T + fuse_b
  gemm(XCAT, 2 * DM, P(P_FUSE_W), P(P_FUSE_B), H, NTOK, DM, 2 * DM, ACT_NONE, 0, stream);

  // ---- fusion MLP blocks (prenorm residual)
  for (int f = 0; f < 2; ++f) {
    int fb = P_FUS0_B1 + f * 6;   // b1,b2,ln_b,ln_w,w1,w2
    ln_kernel<<<(NTOK + 3) / 4, 128, 0, stream>>>(H, P(fb + 3), P(fb + 2), XN, NTOK);
    gemm(XN, DM, P(fb + 4), P(fb + 0), BIG, NTOK, 4 * DM, DM, ACT_GELU, 0, stream);
    gemm(BIG, 4 * DM, P(fb + 5), P(fb + 1), H, NTOK, DM, 4 * DM, ACT_NONE, 1, stream);
  }

  posenc_kernel<<<(NTOK * DM + 255) / 256, 256, 0, stream>>>(H);

  // ---- bidirectional Mamba layers
  for (int L = 0; L < DEPTH; ++L) {
    int lb = P_L0 + L * 17;
    rmsadd_kernel<<<(NTOK + 3) / 4, 128, 0, stream>>>(
        H, RES, P(lb + L_NORMW), XN, NTOK, L == 0);
    gemm(XN, DM, P(lb + L_INW), nullptr, BIG, NTOK, 2 * DI, DM, ACT_NONE, 0, stream);
    for (int dir = 0; dir < 2; ++dir) {
      int g = lb + (dir ? L_bA : L_fA);  // A_log,D,conv_b,conv_w,dt_b,dt_w,xproj_w
      conv_silu_kernel<<<(NTOK * DI + 255) / 256, 256, 0, stream>>>(
          BIG, P(g + 3), P(g + 2), XS, dir);
      gemm(XS, DI, P(g + 6), nullptr, DBL, NTOK, 48, DI, ACT_NONE, 0, stream);
      gemm(DBL, 48, P(g + 5), P(g + 4), DTV, NTOK, DI, 16, ACT_SOFTPLUS, 0, stream);
      scan_kernel<<<BB, DI, 0, stream>>>(XS, DBL, DTV, P(g + 0), P(g + 1),
                                         YACC, dir, dir);
    }
    combine_kernel<<<(NTOK * DI + 255) / 256, 256, 0, stream>>>(YACC, BIG, YC);
    gemm(YC, DI, P(lb + L_OUTW), nullptr, H, NTOK, DM, DI, ACT_NONE, 0, stream);
  }

  // ---- heads
  rmsadd_kernel<<<(NTOK + 3) / 4, 128, 0, stream>>>(H, RES, P(P_NORMF), XN, NTOK, 0);
  gemm(XN, DM, P(P_BEAT_W), P(P_BEAT_B), (float*)d_out, NTOK, 2, DM, ACT_NONE, 0, stream);
  mean_kernel<<<(BB * DM + 255) / 256, 256, 0, stream>>>(XN, TF);
  ln_kernel<<<1, 128, 0, stream>>>(TF, P(P_TEMPO_LNW), P(P_TEMPO_LNB), TFN, BB);
  gemm(TFN, DM, P(P_TEMPO_W), P(P_TEMPO_B), (float*)d_out + NTOK * 2, BB, 300, DM,
       ACT_NONE, 0, stream);
}
